// KVMemoryLayer_4595615007204
// MI455X (gfx1250) — compile-verified, matrix operand order checked by reading, not run on hardware
//
#include <hip/hip_runtime.h>
#include <hip/hip_bf16.h>
#include <math.h>

// ---------------------------------------------------------------------------
// Problem constants (from reference)
// ---------------------------------------------------------------------------
#define D_MODEL   512
#define NUM_SLOTS 200000
#define K_TOP     32
#define TAU       10.0f
#define ALPHA     1.0f
#define EPS_NORM  1e-12f
#define NQ        1024          // B*L = 2*512

// GEMM/topk tiling
#define QTILE     32            // queries per block (two 16-row WMMA A tiles)
#define KCHUNK    2048          // keys per block
#define NCHUNK    98            // ceil(200000/2048) -> covers 200704
#define NQT       (NQ / QTILE)  // 32 query tiles
#define CAND      (NCHUNK * K_TOP)  // 3136 candidates per query
#define KEYS_PER_PASS 256       // 8 compute waves x 32 keys
#define NPASS     (KCHUNK / KEYS_PER_PASS)  // 8

#define QPITCH    520           // bf16 per LDS q row (512 + 8 pad -> conflict-free ds_load_b128)
#define SPITCH    260           // f32 score pitch (256 + 4 pad)

typedef __attribute__((ext_vector_type(16))) __bf16 v16bf;
typedef __attribute__((ext_vector_type(8)))  __bf16 v8bf;
typedef __attribute__((ext_vector_type(8)))  float  v8f;

#define NEG_INF (-3.0e38f)

// ---------------------------------------------------------------------------
// Kernel 1: q = l2norm(x @ W_q^T) -> bf16 ; gate = sigmoid(x . gate_w + b)
// ---------------------------------------------------------------------------
__global__ __launch_bounds__(256) void prep_q(
    const float* __restrict__ x, const float* __restrict__ Wq,
    const float* __restrict__ gate_w, const float* __restrict__ gate_b,
    __bf16* __restrict__ qn, float* __restrict__ g)
{
    const int row = blockIdx.x;      // 0..1023
    const int tid = threadIdx.x;     // 0..255

    __shared__ float xrow[D_MODEL];
    __shared__ float red[256];

    xrow[tid]       = x[(size_t)row * D_MODEL + tid];
    xrow[tid + 256] = x[(size_t)row * D_MODEL + tid + 256];
    __syncthreads();

    const float* w0 = Wq + (size_t)tid * D_MODEL;
    const float* w1 = Wq + (size_t)(tid + 256) * D_MODEL;
    float acc0 = 0.f, acc1 = 0.f;
    #pragma unroll 4
    for (int i = 0; i < D_MODEL; ++i) {
        float xi = xrow[i];
        acc0 += xi * w0[i];
        acc1 += xi * w1[i];
    }

    red[tid] = acc0 * acc0 + acc1 * acc1;
    __syncthreads();
    for (int s = 128; s > 0; s >>= 1) {
        if (tid < s) red[tid] += red[tid + s];
        __syncthreads();
    }
    const float inv = 1.0f / fmaxf(sqrtf(red[0]), EPS_NORM);
    __syncthreads();

    qn[(size_t)row * D_MODEL + tid]       = (__bf16)(acc0 * inv);
    qn[(size_t)row * D_MODEL + tid + 256] = (__bf16)(acc1 * inv);

    red[tid] = xrow[tid] * gate_w[tid] + xrow[tid + 256] * gate_w[tid + 256];
    __syncthreads();
    for (int s = 128; s > 0; s >>= 1) {
        if (tid < s) red[tid] += red[tid + s];
        __syncthreads();
    }
    if (tid == 0) {
        float z = red[0] + gate_b[0];
        g[row] = 1.0f / (1.0f + __expf(-z));
    }
}

// ---------------------------------------------------------------------------
// Kernel 2: inv_norm[m] = 1/max(||keys_m||, eps) * (1/sqrt(D))
// ---------------------------------------------------------------------------
__global__ __launch_bounds__(256) void prep_keys(
    const float* __restrict__ keys, float* __restrict__ inv_norm)
{
    const int lane = threadIdx.x & 31;
    const int row  = blockIdx.x * 8 + (threadIdx.x >> 5);
    if (row >= NUM_SLOTS) return;

    const float* kr = keys + (size_t)row * D_MODEL;
    float s = 0.f;
    #pragma unroll
    for (int j = lane; j < D_MODEL; j += 32) {
        float v = kr[j];
        s += v * v;
    }
    #pragma unroll
    for (int o = 16; o > 0; o >>= 1) s += __shfl_xor(s, o, 32);

    if (lane == 0) {
        const float rsD = 0.044194173824159216f; // 1/sqrt(512)
        inv_norm[row] = (1.0f / fmaxf(sqrtf(s), EPS_NORM)) * rsD;
    }
}

// ---------------------------------------------------------------------------
// Kernel 3: bf16 WMMA scores + overlapped per-chunk top-32
// grid = (32 qtiles [fast], 98 chunks); block = 288 (8 compute waves + 1 topk wave)
// Each compute wave: 32 keys (two B tiles), 4 accumulators; A ds_load feeds 4 WMMAs.
// inv_norm scaling is applied once to the f32 accumulators (per-column constant).
// ---------------------------------------------------------------------------
__global__ __launch_bounds__(288) void scores_topk(
    const float* __restrict__ keys, const __bf16* __restrict__ qn,
    const float* __restrict__ inv_norm,
    float* __restrict__ cand_v, int* __restrict__ cand_i)
{
    const int tid   = threadIdx.x;
    const int lane  = tid & 31;
    const int wave  = tid >> 5;      // 0..8
    const int half  = lane >> 4;     // K-half select for 16-bit fragments
    const int n     = lane & 15;     // A row / B col within 16-tile
    const int qt    = blockIdx.x;    // 0..31
    const int chunk = blockIdx.y;    // 0..97
    const int chunk0 = chunk * KCHUNK;

    __shared__ __bf16 qs[QTILE][QPITCH];          // 33,280 B
    __shared__ float  sc[2][QTILE][SPITCH];       // 66,560 B (double buffer)
    __shared__ float  topv[QTILE][K_TOP];         //  4,096 B
    __shared__ int    topi[QTILE][K_TOP];         //  4,096 B
    __shared__ float  minv[QTILE];
    __shared__ int    minpos[QTILE];

    // ---- stage q tile (bf16) into LDS -------------------------------------
    const __bf16* qbase = qn + (size_t)qt * QTILE * D_MODEL;
    for (int i = tid; i < QTILE * (D_MODEL / 8); i += 288) {
        int r = i >> 6;             // 64 x v8bf per row
        int c = (i & 63) << 3;
        *(v8bf*)&qs[r][c] = *(const v8bf*)(qbase + (size_t)r * D_MODEL + c);
    }
    for (int i = tid; i < QTILE * K_TOP; i += 288)
        ((float*)topv)[i] = NEG_INF;
    if (tid < QTILE) { minv[tid] = NEG_INF; minpos[tid] = 0; }
    __syncthreads();

    // ---- pipeline: compute pass p (waves 0-7) || topk pass p-1 (wave 8) ---
    for (int p = 0; p < NPASS; ++p) {
        if (wave < 8) {
            const int keybase = chunk0 + p * KEYS_PER_PASS;
            const int k0 = keybase + wave * 32 + n;        // B tile 0 column
            const int k1 = k0 + 16;                        // B tile 1 column
            const bool v0 = (k0 < NUM_SLOTS);
            const bool v1 = (k1 < NUM_SLOTS);
            const float scale0 = v0 ? inv_norm[k0] : 0.0f; // folds 1/sqrt(D)
            const float scale1 = v1 ? inv_norm[k1] : 0.0f;
            const float* kp0 = keys + (v0 ? (size_t)k0 * D_MODEL : 0);
            const float* kp1 = keys + (v1 ? (size_t)k1 * D_MODEL : 0);

            v8f c0 = {}, c1 = {}, c2 = {}, c3 = {};
            for (int ks = 0; ks < 16; ++ks) {
                const int kw = ks * 32;
                // --- A fragments from LDS (shared by 4 WMMAs) --------------
                v16bf a0, a1;
                {
                    const v8bf lo0 = *(const v8bf*)&qs[n][kw + 8 * half];
                    const v8bf hi0 = *(const v8bf*)&qs[n][kw + 16 + 8 * half];
                    const v8bf lo1 = *(const v8bf*)&qs[16 + n][kw + 8 * half];
                    const v8bf hi1 = *(const v8bf*)&qs[16 + n][kw + 16 + 8 * half];
                    #pragma unroll
                    for (int e = 0; e < 8; ++e) {
                        a0[e] = lo0[e]; a0[e + 8] = hi0[e];
                        a1[e] = lo1[e]; a1[e + 8] = hi1[e];
                    }
                }
                // --- B fragments: raw fp32 keys -> bf16 (no per-elem scale)
                v16bf b0, b1;
                {
                    const float4 f0 = *(const float4*)(kp0 + kw + 8 * half);
                    const float4 f1 = *(const float4*)(kp0 + kw + 8 * half + 4);
                    const float4 f2 = *(const float4*)(kp0 + kw + 16 + 8 * half);
                    const float4 f3 = *(const float4*)(kp0 + kw + 16 + 8 * half + 4);
                    b0[0]  = (__bf16)f0.x; b0[1]  = (__bf16)f0.y;
                    b0[2]  = (__bf16)f0.z; b0[3]  = (__bf16)f0.w;
                    b0[4]  = (__bf16)f1.x; b0[5]  = (__bf16)f1.y;
                    b0[6]  = (__bf16)f1.z; b0[7]  = (__bf16)f1.w;
                    b0[8]  = (__bf16)f2.x; b0[9]  = (__bf16)f2.y;
                    b0[10] = (__bf16)f2.z; b0[11] = (__bf16)f2.w;
                    b0[12] = (__bf16)f3.x; b0[13] = (__bf16)f3.y;
                    b0[14] = (__bf16)f3.z; b0[15] = (__bf16)f3.w;
                }
                {
                    const float4 f0 = *(const float4*)(kp1 + kw + 8 * half);
                    const float4 f1 = *(const float4*)(kp1 + kw + 8 * half + 4);
                    const float4 f2 = *(const float4*)(kp1 + kw + 16 + 8 * half);
                    const float4 f3 = *(const float4*)(kp1 + kw + 16 + 8 * half + 4);
                    b1[0]  = (__bf16)f0.x; b1[1]  = (__bf16)f0.y;
                    b1[2]  = (__bf16)f0.z; b1[3]  = (__bf16)f0.w;
                    b1[4]  = (__bf16)f1.x; b1[5]  = (__bf16)f1.y;
                    b1[6]  = (__bf16)f1.z; b1[7]  = (__bf16)f1.w;
                    b1[8]  = (__bf16)f2.x; b1[9]  = (__bf16)f2.y;
                    b1[10] = (__bf16)f2.z; b1[11] = (__bf16)f2.w;
                    b1[12] = (__bf16)f3.x; b1[13] = (__bf16)f3.y;
                    b1[14] = (__bf16)f3.z; b1[15] = (__bf16)f3.w;
                }
                c0 = __builtin_amdgcn_wmma_f32_16x16x32_bf16(
                         false, a0, false, b0, (short)0, c0, false, false);
                c1 = __builtin_amdgcn_wmma_f32_16x16x32_bf16(
                         false, a1, false, b0, (short)0, c1, false, false);
                c2 = __builtin_amdgcn_wmma_f32_16x16x32_bf16(
                         false, a0, false, b1, (short)0, c2, false, false);
                c3 = __builtin_amdgcn_wmma_f32_16x16x32_bf16(
                         false, a1, false, b1, (short)0, c3, false, false);
            }

            // ---- scale once per tile, spill to double-buffered LDS --------
            float (*scb)[SPITCH] = sc[p & 1];
            const int col0 = wave * 32 + n;
            #pragma unroll
            for (int r = 0; r < 8; ++r) {
                scb[r + 8 * half][col0]           = c0[r] * scale0;
                scb[16 + r + 8 * half][col0]      = c1[r] * scale0;
                scb[r + 8 * half][col0 + 16]      = c2[r] * scale1;
                scb[16 + r + 8 * half][col0 + 16] = c3[r] * scale1;
            }
        } else if (p > 0) {
            // ---- top-k scan of previous pass (overlaps compute) -----------
            const int q = lane;
            const int keybase = chunk0 + (p - 1) * KEYS_PER_PASS;
            const float (*scb)[SPITCH] = sc[(p - 1) & 1];
            float mv = minv[q]; int mp = minpos[q];
            for (int j = 0; j < KEYS_PER_PASS; ++j) {
                int key = keybase + j;
                float v = scb[q][j];
                if (key < NUM_SLOTS && v > mv) {
                    topv[q][mp] = v; topi[q][mp] = key;
                    mv = topv[q][0]; mp = 0;
                    #pragma unroll
                    for (int t = 1; t < K_TOP; ++t) {
                        float tv = topv[q][t];
                        if (tv < mv) { mv = tv; mp = t; }
                    }
                }
            }
            minv[q] = mv; minpos[q] = mp;
        }
        __syncthreads();
    }

    // ---- drain: top-k for the final pass ----------------------------------
    if (wave == 8) {
        const int q = lane;
        const int keybase = chunk0 + (NPASS - 1) * KEYS_PER_PASS;
        const float (*scb)[SPITCH] = sc[(NPASS - 1) & 1];
        float mv = minv[q]; int mp = minpos[q];
        for (int j = 0; j < KEYS_PER_PASS; ++j) {
            int key = keybase + j;
            float v = scb[q][j];
            if (key < NUM_SLOTS && v > mv) {
                topv[q][mp] = v; topi[q][mp] = key;
                mv = topv[q][0]; mp = 0;
                #pragma unroll
                for (int t = 1; t < K_TOP; ++t) {
                    float tv = topv[q][t];
                    if (tv < mv) { mv = tv; mp = t; }
                }
            }
        }
    }
    __syncthreads();

    // ---- emit per-chunk candidates ----------------------------------------
    for (int i = tid; i < QTILE * K_TOP; i += 288) {
        int q = i >> 5, k = i & 31;
        size_t base = (((size_t)(qt * QTILE + q)) * NCHUNK + chunk) * K_TOP + k;
        cand_v[base] = topv[q][k];
        cand_i[base] = topi[q][k];
    }
}

// ---------------------------------------------------------------------------
// Kernel 4: merge candidates -> exact top-32, softmax, weighted val gather
// ---------------------------------------------------------------------------
__global__ __launch_bounds__(256) void merge_out(
    const float* __restrict__ cand_v, const int* __restrict__ cand_i,
    const float* __restrict__ vals, const float* __restrict__ g,
    float* __restrict__ out)
{
    const int q   = blockIdx.x;
    const int tid = threadIdx.x;

    __shared__ float cv[CAND];
    __shared__ int   ci[CAND];
    __shared__ float rv[256];
    __shared__ int   rp[256];
    __shared__ float selv[K_TOP];
    __shared__ int   seli[K_TOP];
    __shared__ float attn[K_TOP];

    const float* cvq = cand_v + (size_t)q * CAND;
    const int*   ciq = cand_i + (size_t)q * CAND;
    for (int i = tid; i < CAND; i += 256) { cv[i] = cvq[i]; ci[i] = ciq[i]; }
    __syncthreads();

    for (int it = 0; it < K_TOP; ++it) {
        float bm = NEG_INF; int bp = 0;
        for (int j = tid; j < CAND; j += 256) {
            float v = cv[j];
            if (v > bm) { bm = v; bp = j; }
        }
        rv[tid] = bm; rp[tid] = bp;
        __syncthreads();
        for (int s = 128; s > 0; s >>= 1) {
            if (tid < s && rv[tid + s] > rv[tid]) {
                rv[tid] = rv[tid + s]; rp[tid] = rp[tid + s];
            }
            __syncthreads();
        }
        if (tid == 0) {
            selv[it] = rv[0];
            seli[it] = ci[rp[0]];
            cv[rp[0]] = NEG_INF;
        }
        __syncthreads();
    }

    if (tid < K_TOP) {
        float v = selv[tid] / TAU;
        float m = v;
        #pragma unroll
        for (int o = 16; o > 0; o >>= 1) m = fmaxf(m, __shfl_xor(m, o, 32));
        float e = __expf(v - m);
        float s = e;
        #pragma unroll
        for (int o = 16; o > 0; o >>= 1) s += __shfl_xor(s, o, 32);
        attn[tid] = e / s;
    }
    __syncthreads();

    const float gq = ALPHA * g[q];
    for (int d = tid; d < D_MODEL; d += 256) {
        float acc = 0.f;
        #pragma unroll 8
        for (int k = 0; k < K_TOP; ++k)
            acc += attn[k] * vals[(size_t)seli[k] * D_MODEL + d];
        out[(size_t)q * D_MODEL + d] = gq * acc;
    }
}

// ---------------------------------------------------------------------------
// launch
// ---------------------------------------------------------------------------
extern "C" void kernel_launch(void* const* d_in, const int* in_sizes, int n_in,
                              void* d_out, int out_size, void* d_ws, size_t ws_size,
                              hipStream_t stream)
{
    const float* x      = (const float*)d_in[0];  // [2,512,512]
    const float* keys   = (const float*)d_in[1];  // [200000,512]
    const float* vals   = (const float*)d_in[2];  // [200000,512]
    const float* Wq     = (const float*)d_in[3];  // [512,512]
    const float* gate_w = (const float*)d_in[4];  // [512]
    const float* gate_b = (const float*)d_in[5];  // [1]
    float* out = (float*)d_out;                   // [2,512,512] fp32

    char* w = (char*)d_ws;
    size_t off = 0;
    __bf16* qn = (__bf16*)(w + off);  off += (size_t)NQ * D_MODEL * 2;       // 1 MB
    float* g   = (float*)(w + off);   off += (size_t)NQ * 4;                 // 4 KB
    float* inv = (float*)(w + off);   off += ((size_t)NUM_SLOTS * 4 + 255) & ~(size_t)255;
    float* cand_v = (float*)(w + off); off += (size_t)NQ * CAND * 4;         // 12.85 MB
    int*   cand_i = (int*)(w + off);   off += (size_t)NQ * CAND * 4;         // 12.85 MB
    (void)off; (void)ws_size; (void)in_sizes; (void)n_in; (void)out_size;

    prep_q<<<NQ, 256, 0, stream>>>(x, Wq, gate_w, gate_b, qn, g);
    prep_keys<<<(NUM_SLOTS + 7) / 8, 256, 0, stream>>>(keys, inv);
    scores_topk<<<dim3(NQT, NCHUNK), 288, 0, stream>>>(keys, qn, inv, cand_v, cand_i);
    merge_out<<<NQ, 256, 0, stream>>>(cand_v, cand_i, vals, g, out);
}
